// finetunemodel_18176301597207
// MI455X (gfx1250) — matrix operand order repaired and checked by reading, not compiled, not run on hardware
//
#include <hip/hip_runtime.h>
#include <hip/hip_bf16.h>

typedef _Float16 half_t;
typedef __attribute__((ext_vector_type(16))) _Float16 v16h;
typedef __attribute__((ext_vector_type(8)))  _Float16 v8h;
typedef __attribute__((ext_vector_type(8)))  float    v8f;

#define HW      600
#define EMB     256
#define CH      64
#define MAXB    100
#define KSZ     7
#define NLAYER  7
#define CROPN   50
#define NTRI    500   // HW - 2*CROP

// ---------------------------------------------------------------------------
// A-fragment gather for v_wmma_f32_16x16x32_f16 (16-bit A, 16x32):
//  lane = l, m = l&15, hi = l>>4. halves 0..7 -> K = c0..c0+7,
//  halves 8..15 -> K = c0+16..c0+23, with c0 = kchunk*32 + (hi?8:0).
// Both 8-half groups are contiguous & 16B aligned -> two b128 loads.
// ---------------------------------------------------------------------------
__device__ inline v16h frag_a_from(const half_t* p, int c0) {
  v8h lo = *(const v8h*)(p + c0);
  v8h hi = *(const v8h*)(p + c0 + 16);
  v16h r;
#pragma unroll
  for (int e = 0; e < 8; ++e) { r[e] = lo[e]; r[e + 8] = hi[e]; }
  return r;
}

// ===========================================================================
// Prep kernels (run once per launch, tiny)
// ===========================================================================
__global__ void prep_win_kernel(const float* __restrict__ w, half_t* __restrict__ wf) {
  int idx = blockIdx.x * blockDim.x + threadIdx.x;
  if (idx < CH * EMB) wf[idx] = (half_t)w[idx];   // [o][c], layout already matches
}

// src w_dil: [layer][o][c][ky][kx]  ->  dst: [layer][ky*7+kx][o][c]  (f16)
__global__ void prep_wdil_kernel(const float* __restrict__ w, half_t* __restrict__ wf) {
  int idx = blockIdx.x * blockDim.x + threadIdx.x;
  const int total = NLAYER * KSZ * KSZ * CH * CH;
  if (idx >= total) return;
  int c = idx & 63;
  int o = (idx >> 6) & 63;
  int rest = idx >> 12;          // l*49 + t
  int t = rest % 49;
  int l = rest / 49;
  int ky = t / KSZ, kx = t % KSZ;
  wf[idx] = (half_t)w[(((l * CH + o) * CH + c) * KSZ + ky) * KSZ + kx];
}

// fold conv bias + BN into per-channel scale/shift:
//   y = z*scale + shift,  scale = gamma*rsqrt(var+eps), shift = (b-mean)*scale+beta
__global__ void prep_bn_kernel(const float* __restrict__ bdil, const float* __restrict__ g,
                               const float* __restrict__ be, const float* __restrict__ mu,
                               const float* __restrict__ var,
                               float* __restrict__ scale, float* __restrict__ shift) {
  int idx = blockIdx.x * blockDim.x + threadIdx.x;
  if (idx >= NLAYER * CH) return;
  float s = g[idx] * rsqrtf(var[idx] + 1e-5f);
  scale[idx] = s;
  shift[idx] = (bdil[idx] - mu[idx]) * s + be[idx];
}

// fold out-conv into head: v[c] = sum_o w_head[o]*w_out[o][c]; v[64] = s0
__global__ void prep_head_kernel(const float* __restrict__ w_out, const float* __restrict__ b_out,
                                 const float* __restrict__ w_head, const float* __restrict__ b_head,
                                 float* __restrict__ v) {
  int c = threadIdx.x;
  if (c < CH) {
    float s = 0.f;
    for (int o = 0; o < CH; ++o) s += w_head[o] * w_out[o * CH + c];
    v[c] = s;
  } else if (c == CH) {
    float s = b_head[0];
    for (int o = 0; o < CH; ++o) s += w_head[o] * b_out[o];
    v[CH] = s;
  }
}

// ===========================================================================
// Stage 1: pair features fused into the 256->64 1x1 conv (implicit GEMM).
// Block = 128 thr (4 waves), covers 64 pixels of one row i.
// LDS-stage pair (64px x 256ch, f16), then each wave: 16x64x256 WMMA GEMM.
// ===========================================================================
__global__ __launch_bounds__(128)
void pair_inconv_kernel(const float* __restrict__ x, const float* __restrict__ dist,
                        const half_t* __restrict__ winf, const float* __restrict__ b_in,
                        half_t* __restrict__ h0) {
  __shared__ half_t lds[64 * EMB];   // 32 KB
  const int i   = blockIdx.y;
  const int j0b = blockIdx.x * 64;
  const int tid = threadIdx.x;

  for (int idx = tid; idx < 64 * EMB; idx += 128) {
    int p = idx >> 8;
    int c = idx & (EMB - 1);
    int j = j0b + p;
    float v = 0.f;
    if (j < HW) {
      float xi = x[i * EMB + c];
      float xj = x[j * EMB + c];
      int ad = i - j; if (ad < 0) ad = -ad; if (ad > MAXB) ad = MAXB;
      v = 0.5f * (xi + xj) + xi * xj + dist[ad * EMB + c];
    }
    lds[idx] = (half_t)v;
  }
  __syncthreads();

  const int lane = tid & 31;
  const int wave = tid >> 5;
  const int m    = lane & 15;
  const int hi   = lane >> 4;

  const half_t* arow = &lds[(wave * 16 + m) * EMB];

  v8f zero = {};
  v8f acc[4] = {zero, zero, zero, zero};

#pragma unroll
  for (int kc = 0; kc < 8; ++kc) {               // K = 256 in chunks of 32
    v16h a = frag_a_from(arow, kc * 32 + (hi ? 8 : 0));
#pragma unroll
    for (int nt = 0; nt < 4; ++nt) {
      int o = nt * 16 + m;                       // B: N = lane&15
      v16h b = frag_a_from(winf + o * EMB, kc * 32 + hi * 8);
      acc[nt] = __builtin_amdgcn_wmma_f32_16x16x32_f16(
          false, a, false, b, (short)0, acc[nt], false, false);
    }
  }

  const int mb = hi * 8;
#pragma unroll
  for (int nt = 0; nt < 4; ++nt) {
    int ch = nt * 16 + m;
    float bb = b_in[ch];
#pragma unroll
    for (int r = 0; r < 8; ++r) {
      int jj = j0b + wave * 16 + mb + r;
      if (jj < HW)
        h0[((long)i * HW + jj) * CH + ch] = (half_t)(acc[nt][r] + bb);
    }
  }
}

// ===========================================================================
// Stage 2: dilated 7x7 conv, 64->64 ch, as implicit GEMM per tap.
// Block = 128 thr (4 waves); each wave owns 32 output pixels x 64 channels
// (2 M-tiles x 4 N-tiles of f32 accumulators = 64 VGPRs).
// Per tap: 4 A-frags (2 M-tiles x K=64), 8 B-frags reused across both
// M-tiles, 16 x v_wmma_f32_16x16x32_f16  ->  1.5 vmem issues per WMMA.
// Epilogue: relu(acc*scale + shift + residual) -> f16.
// ===========================================================================
__global__ __launch_bounds__(128)
void dilconv_kernel(const half_t* __restrict__ hin, half_t* __restrict__ hout,
                    const half_t* __restrict__ wl,      // [49][64][64] f16
                    const float* __restrict__ scale, const float* __restrict__ shift,
                    int d) {
  const int i    = blockIdx.y;
  const int lane = threadIdx.x & 31;
  const int wave = threadIdx.x >> 5;
  const int j0   = blockIdx.x * 128 + wave * 32;   // 32 pixels per wave
  if (j0 >= HW) return;                            // wave-uniform
  const int m  = lane & 15;
  const int hi = lane >> 4;

  v8f zero = {};
  v8f acc[8] = {zero, zero, zero, zero, zero, zero, zero, zero}; // [mt*4+nt]
  v16h zf = {};

  for (int ky = 0; ky < KSZ; ++ky) {
    int ii = i + (ky - 3) * d;
    if (ii < 0 || ii >= HW) continue;              // uniform skip
    for (int kx = 0; kx < KSZ; ++kx) {
      int jb = j0 + (kx - 3) * d;
      if (jb + 31 < 0 || jb >= HW) continue;       // uniform skip (32-px window)

      // A fragments: a[mt*2 + kc], K = ch kc*32 .. kc*32+31
      v16h a[4] = {zf, zf, zf, zf};
#pragma unroll
      for (int mt = 0; mt < 2; ++mt) {
        int jj = jb + mt * 16 + m;
        if (jj >= 0 && jj < HW) {
          const half_t* ap = hin + ((long)ii * HW + jj) * CH;
          a[mt * 2 + 0] = frag_a_from(ap, (hi ? 8 : 0));
          a[mt * 2 + 1] = frag_a_from(ap, 32 + (hi ? 8 : 0));
        }
      }

      const half_t* wp = wl + (long)(ky * KSZ + kx) * CH * CH;
#pragma unroll
      for (int nt = 0; nt < 4; ++nt) {
        int o = nt * 16 + m;
        v16h b0 = frag_a_from(wp + o * CH, hi * 8);
        v16h b1 = frag_a_from(wp + o * CH, 32 + hi * 8);
#pragma unroll
        for (int mt = 0; mt < 2; ++mt) {
          acc[mt * 4 + nt] = __builtin_amdgcn_wmma_f32_16x16x32_f16(
              false, a[mt * 2 + 0], false, b0, (short)0, acc[mt * 4 + nt], false, false);
          acc[mt * 4 + nt] = __builtin_amdgcn_wmma_f32_16x16x32_f16(
              false, a[mt * 2 + 1], false, b1, (short)0, acc[mt * 4 + nt], false, false);
        }
      }
    }
  }

  const int mb = hi * 8;
#pragma unroll
  for (int mt = 0; mt < 2; ++mt) {
#pragma unroll
    for (int nt = 0; nt < 4; ++nt) {
      int ch = nt * 16 + m;
      float sc = scale[ch], sh = shift[ch];
#pragma unroll
      for (int r = 0; r < 8; ++r) {
        int jj = j0 + mt * 16 + mb + r;
        if (jj < HW) {
          long off = ((long)i * HW + jj) * CH + ch;
          float y = acc[mt * 4 + nt][r] * sc + sh + (float)hin[off];
          hout[off] = (half_t)(y > 0.f ? y : 0.f);
        }
      }
    }
  }
}

// ===========================================================================
// Stage 3: crop + upper-tri gather + (folded out-conv ∘ head)
// ===========================================================================
__global__ __launch_bounds__(256)
void head_kernel(const half_t* __restrict__ h, const float* __restrict__ v,
                 float* __restrict__ out) {
  int c = blockIdx.x * 16 + (threadIdx.x & 15);
  int r = blockIdx.y * 16 + (threadIdx.x >> 4);
  if (r >= NTRI || c >= NTRI || c < r) return;
  const half_t* p = h + ((long)(r + CROPN) * HW + (c + CROPN)) * CH;
  float acc = v[CH];
#pragma unroll
  for (int ch = 0; ch < CH; ++ch) acc += v[ch] * (float)p[ch];
  long t = (long)r * NTRI - (long)r * (r - 1) / 2 + (c - r);
  out[t] = acc;
}

// ===========================================================================
extern "C" void kernel_launch(void* const* d_in, const int* in_sizes, int n_in,
                              void* d_out, int out_size, void* d_ws, size_t ws_size,
                              hipStream_t stream) {
  const float* x      = (const float*)d_in[0];
  const float* dist   = (const float*)d_in[1];
  const float* w_in   = (const float*)d_in[2];
  const float* b_in   = (const float*)d_in[3];
  const float* w_dil  = (const float*)d_in[4];
  const float* b_dil  = (const float*)d_in[5];
  const float* gamma  = (const float*)d_in[6];
  const float* beta   = (const float*)d_in[7];
  const float* mean   = (const float*)d_in[8];
  const float* var    = (const float*)d_in[9];
  const float* w_out  = (const float*)d_in[10];
  const float* b_out  = (const float*)d_in[11];
  const float* w_head = (const float*)d_in[12];
  const float* b_head = (const float*)d_in[13];
  float* out = (float*)d_out;

  char* ws = (char*)d_ws;
  size_t off = 0;
  auto carve = [&](size_t bytes) {
    char* p = ws + off;
    off = (off + bytes + 255) & ~(size_t)255;
    return p;
  };

  const size_t NPIX = (size_t)HW * HW;
  half_t* hA    = (half_t*)carve(NPIX * CH * sizeof(half_t));
  half_t* hB    = (half_t*)carve(NPIX * CH * sizeof(half_t));
  half_t* winf  = (half_t*)carve((size_t)CH * EMB * sizeof(half_t));
  half_t* wdilf = (half_t*)carve((size_t)NLAYER * 49 * CH * CH * sizeof(half_t));
  float*  scale = (float*)carve(NLAYER * CH * sizeof(float));
  float*  shift = (float*)carve(NLAYER * CH * sizeof(float));
  float*  vhead = (float*)carve((CH + 1) * sizeof(float));
  (void)in_sizes; (void)n_in; (void)out_size; (void)ws_size;

  // --- prep ---
  prep_win_kernel<<<(CH * EMB + 255) / 256, 256, 0, stream>>>(w_in, winf);
  {
    const int tot = NLAYER * 49 * CH * CH;
    prep_wdil_kernel<<<(tot + 255) / 256, 256, 0, stream>>>(w_dil, wdilf);
  }
  prep_bn_kernel<<<(NLAYER * CH + 255) / 256, 256, 0, stream>>>(
      b_dil, gamma, beta, mean, var, scale, shift);
  prep_head_kernel<<<1, 128, 0, stream>>>(w_out, b_out, w_head, b_head, vhead);

  // --- stage 1: pair + 1x1 in-conv (WMMA) ---
  pair_inconv_kernel<<<dim3((HW + 63) / 64, HW), 128, 0, stream>>>(
      x, dist, winf, b_in, hA);

  // --- stage 2: 7 dilated conv layers (WMMA implicit GEMM), ping-pong ---
  half_t* cur = hA;
  half_t* nxt = hB;
  for (int l = 0; l < NLAYER; ++l) {
    dilconv_kernel<<<dim3((HW + 127) / 128, HW), 128, 0, stream>>>(
        cur, nxt, wdilf + (size_t)l * 49 * CH * CH, scale + l * CH, shift + l * CH, 1 << l);
    half_t* t = cur; cur = nxt; nxt = t;
  }

  // --- stage 3: crop + triu + folded head ---
  head_kernel<<<dim3((NTRI + 15) / 16, (NTRI + 15) / 16), 256, 0, stream>>>(
      cur, vhead, out);
}
